// SpectralConvND_39436389712375
// MI455X (gfx1250) — compile-verified, matrix operand order checked by reading, not run on hardware
//
#include <hip/hip_runtime.h>
#include <math.h>

// ---------------------------------------------------------------------------
// SpectralConv (FNO-style) for MI455X / gfx1250, compile-only tuned.
//
// Pipeline (all truncated dense DFTs, no full FFT):
//   K1: Wt[o,c,i,j]   = sum_r proj[o,r] * coeff[r,c,i,j]          (f32 VALU)
//   K2: Y[b,c,h,j]    = sum_w x * e^{-2pi i j w/256}, j<32        (bf16 WMMA)
//   K3: X[b,c,i,j]    = sum_h Y * e^{-2pi i i h/256}, i<32        (f32 + LDS)
//   K4: G[b,o,i,j]    = sum_c X * Wt            (complex mix)     (f32 VALU)
//   K5: T[b,o,h,j]    = a_j/65536 * sum_i G * e^{+2pi i i h/256}  (f32 VALU)
//   K6: out[b,o,h,w]  = sum_j Tre*cos - Tim*sin                   (bf16 WMMA)
// ---------------------------------------------------------------------------

typedef __attribute__((ext_vector_type(16))) __bf16 v16bf;
typedef __attribute__((ext_vector_type(8)))  float  v8f;
typedef __attribute__((ext_vector_type(4)))  float  f32x4;

#define PI2_256 0.024543692602170255f  // 2*pi/256

// ---- workspace byte offsets ------------------------------------------------
static constexpr size_t OFF_TWH_COS  = 0;                       // 32x256 f32
static constexpr size_t OFF_TWH_SIN  = 32768;
static constexpr size_t OFF_FWDB_COS = 65536;                   // [2][8][32][16] bf16
static constexpr size_t OFF_FWDB_SIN = 81920;                   // stores -sin
static constexpr size_t OFF_INVB_COS = 98304;                   // [16][32][16] bf16
static constexpr size_t OFF_INVB_MSIN= 114688;                  // stores -sin
static constexpr size_t OFF_WT_RE    = 131072;                  // 32^4 f32
static constexpr size_t OFF_WT_IM    = OFF_WT_RE + 4194304;
static constexpr size_t OFF_Y_RE     = OFF_WT_IM + 4194304;     // 131072x32 f32 (aliased by T)
static constexpr size_t OFF_Y_IM     = OFF_Y_RE + 16777216;
static constexpr size_t OFF_X_RE     = OFF_Y_IM + 16777216;     // 16*32*32*32 f32
static constexpr size_t OFF_X_IM     = OFF_X_RE + 2097152;
static constexpr size_t OFF_G_RE     = OFF_X_IM + 2097152;
static constexpr size_t OFF_G_IM     = OFF_G_RE + 2097152;      // total ~50.5 MB

__device__ __forceinline__ v8f wmma_bf16(v16bf a, v16bf b, v8f c) {
  return __builtin_amdgcn_wmma_f32_16x16x32_bf16(false, a, false, b, (short)0, c,
                                                 false, false);
}

// ---- K0a: h-axis twiddles (f32, 32 x 256) ----------------------------------
__global__ void k_build_twh(float* __restrict__ twc, float* __restrict__ tws) {
  int t = blockIdx.x * 256 + threadIdx.x;    // 8192
  int i = t >> 8, h = t & 255;
  float ang = (float)((i * h) & 255) * PI2_256;
  twc[t] = cosf(ang);
  tws[t] = sinf(ang);
}

// ---- K0b: forward-w bf16 B fragments, pre-swizzled to WMMA lane layout -----
// B mirrors the 16-bit A layout (§7.12.2): lane<16 holds K {kb..kb+7,kb+16..23},
// lane>=16 holds K {kb+8..15, kb+24..31}; N = lane&15.
__global__ void k_build_fwdB(__bf16* __restrict__ bcos, __bf16* __restrict__ bsin) {
  int t = blockIdx.x * 256 + threadIdx.x;    // 8192 = 2*8*32*16
  int e = t & 15, lane = (t >> 4) & 31, kit = (t >> 9) & 7, nt = t >> 12;
  int j = nt * 16 + (lane & 15);
  int khalf = (lane >> 4) * 8;
  int k = kit * 32 + ((e < 8) ? (khalf + e) : (16 + khalf + (e - 8)));
  float ang = (float)((j * k) & 255) * PI2_256;
  bcos[t] = (__bf16)cosf(ang);
  bsin[t] = (__bf16)(-sinf(ang));            // e^{-i theta}: Yim accumulates -sin
}

// ---- K0c: inverse-w bf16 B fragments (16 N-tiles, K=32=j) ------------------
__global__ void k_build_invB(__bf16* __restrict__ bcos, __bf16* __restrict__ bmsin) {
  int t = blockIdx.x * 256 + threadIdx.x;    // 8192 = 16*32*16
  int e = t & 15, lane = (t >> 4) & 31, nt = t >> 9;
  int w = nt * 16 + (lane & 15);
  int khalf = (lane >> 4) * 8;
  int j = (e < 8) ? (khalf + e) : (16 + khalf + (e - 8));
  float ang = (float)((j * w) & 255) * PI2_256;
  bcos[t]  = (__bf16)cosf(ang);
  bmsin[t] = (__bf16)(-sinf(ang));           // out = Re(T e^{+i}) = Tre*c - Tim*s
}

// ---- K1: Wt = proj @ coeff (over r), complex halves ------------------------
__global__ void k_wt(const float* __restrict__ proj,
                     const float* __restrict__ cre, const float* __restrict__ cim,
                     float* __restrict__ wre, float* __restrict__ wim) {
  int t = blockIdx.x * 256 + threadIdx.x;    // 1048576 = 32^4, layout [o][c][i][j]
  int base = t & 32767;
  int o = t >> 15;
  float ar = 0.f, ai = 0.f;
#pragma unroll 8
  for (int r = 0; r < 32; ++r) {
    float p = proj[o * 32 + r];
    ar += p * cre[(size_t)r * 32768 + base];
    ai += p * cim[(size_t)r * 32768 + base];
  }
  wre[t] = ar;
  wim[t] = ai;
}

// ---- K2: forward w-DFT via WMMA bf16. One wave per 16-row tile. ------------
// rows = (b*32+c)*256+h = 131072; A = x rows (K=256), B = twiddles (32 cols).
__global__ __launch_bounds__(256) void k_fwd_wdft(
    const float* __restrict__ x, float* __restrict__ yre, float* __restrict__ yim,
    const __bf16* __restrict__ bcos, const __bf16* __restrict__ bsin) {
  const int lane = threadIdx.x & 31;
  const int wave = (blockIdx.x << 3) + (threadIdx.x >> 5);
  const int r0 = wave << 4;
  const int lr = lane & 15, hi = lane >> 4;
  const float* rowp = x + (size_t)(r0 + lr) * 256 + hi * 8;
  v8f aC0 = {0,0,0,0,0,0,0,0}, aS0 = {0,0,0,0,0,0,0,0};
  v8f aC1 = {0,0,0,0,0,0,0,0}, aS1 = {0,0,0,0,0,0,0,0};
#pragma unroll
  for (int kit = 0; kit < 8; ++kit) {
    const float* p = rowp + kit * 32;
    __builtin_prefetch(p + 64, 0, 1);        // global_prefetch_b8 K chunk two iterations ahead
    f32x4 p0 = *(const f32x4*)(p);
    f32x4 p1 = *(const f32x4*)(p + 4);
    f32x4 p2 = *(const f32x4*)(p + 16);
    f32x4 p3 = *(const f32x4*)(p + 20);
    v16bf a;
#pragma unroll
    for (int e = 0; e < 4; ++e) {
      a[e]      = (__bf16)p0[e];
      a[e + 4]  = (__bf16)p1[e];
      a[e + 8]  = (__bf16)p2[e];
      a[e + 12] = (__bf16)p3[e];
    }
    const v16bf bc0 = *(const v16bf*)(bcos + (((0 * 8 + kit) * 32 + lane) << 4));
    const v16bf bs0 = *(const v16bf*)(bsin + (((0 * 8 + kit) * 32 + lane) << 4));
    const v16bf bc1 = *(const v16bf*)(bcos + (((1 * 8 + kit) * 32 + lane) << 4));
    const v16bf bs1 = *(const v16bf*)(bsin + (((1 * 8 + kit) * 32 + lane) << 4));
    aC0 = wmma_bf16(a, bc0, aC0);
    aS0 = wmma_bf16(a, bs0, aS0);
    aC1 = wmma_bf16(a, bc1, aC1);
    aS1 = wmma_bf16(a, bs1, aS1);
  }
  const int col = lane & 15;
#pragma unroll
  for (int r = 0; r < 8; ++r) {              // D layout: vgpr r -> row r + hi*8
    size_t o = (size_t)(r0 + r + hi * 8) * 32 + col;
    yre[o] = aC0[r]; yre[o + 16] = aC1[r];
    yim[o] = aS0[r]; yim[o + 16] = aS1[r];
  }
}

// ---- K3: forward h-DFT, Y tile staged through LDS (32x reuse) --------------
__global__ __launch_bounds__(256) void k_fwd_hdft(
    const float* __restrict__ yre, const float* __restrict__ yim,
    float* __restrict__ xre, float* __restrict__ xim,
    const float* __restrict__ twc, const float* __restrict__ tws) {
  __shared__ float sre[8192];
  __shared__ float sim[8192];
  const int bc = blockIdx.x;                 // (b*32+c), 512 blocks
  const size_t base = (size_t)bc * 8192;
  for (int t = threadIdx.x; t < 8192; t += 256) {
    sre[t] = yre[base + t];
    sim[t] = yim[base + t];
  }
  __syncthreads();
  const int j = threadIdx.x & 31;
  const int i0 = threadIdx.x >> 5;
#pragma unroll
  for (int ii = 0; ii < 4; ++ii) {
    const int i = i0 + ii * 8;
    const float* tc = twc + i * 256;
    const float* ts = tws + i * 256;
    float xr = 0.f, xi = 0.f;
#pragma unroll 4
    for (int h = 0; h < 256; ++h) {          // e^{-i}: (c - i s)
      float c = tc[h], s = ts[h];
      float yr = sre[h * 32 + j], yq = sim[h * 32 + j];
      xr += yr * c + yq * s;
      xi += yq * c - yr * s;
    }
    size_t o = (size_t)bc * 1024 + i * 32 + j;
    xre[o] = xr; xim[o] = xi;
  }
}

// ---- K4: per-frequency complex channel mix (L2-resident) -------------------
__global__ void k_chanmix(const float* __restrict__ xre, const float* __restrict__ xim,
                          const float* __restrict__ wre, const float* __restrict__ wim,
                          float* __restrict__ gre, float* __restrict__ gim) {
  int t = blockIdx.x * 256 + threadIdx.x;    // 524288 = b,o,i,j
  int ij = t & 1023;
  int o = (t >> 10) & 31;
  int b = t >> 15;
  const float* xr = xre + (size_t)b * 32768 + ij;
  const float* xq = xim + (size_t)b * 32768 + ij;
  const float* wr = wre + (size_t)o * 32768 + ij;
  const float* wq = wim + (size_t)o * 32768 + ij;
  float gr = 0.f, gi = 0.f;
#pragma unroll 4
  for (int c = 0; c < 32; ++c) {
    float a = xr[c * 1024], bb = xq[c * 1024];
    float cc = wr[c * 1024], dd = wq[c * 1024];
    gr += a * cc - bb * dd;
    gi += a * dd + bb * cc;
  }
  size_t og = (size_t)(b * 32 + o) * 1024 + ij;
  gre[og] = gr; gim[og] = gi;
}

// ---- K5: inverse h-DFT (K=32) with irfft hermitian scale folded in ---------
__global__ void k_inv_hdft(const float* __restrict__ gre, const float* __restrict__ gim,
                           float* __restrict__ tre, float* __restrict__ tim,
                           const float* __restrict__ twc, const float* __restrict__ tws) {
  int t = blockIdx.x * 256 + threadIdx.x;    // 4194304 = bo,h,j
  int j = t & 31;
  int h = (t >> 5) & 255;
  int bo = t >> 13;
  const float* gr = gre + (size_t)bo * 1024 + j;
  const float* gi = gim + (size_t)bo * 1024 + j;
  float sr = 0.f, si = 0.f;
#pragma unroll 8
  for (int i = 0; i < 32; ++i) {             // e^{+i}: (c + i s)
    float c = twc[i * 256 + h], s = tws[i * 256 + h];
    float a = gr[i * 32], bb = gi[i * 32];
    sr += a * c - bb * s;
    si += a * s + bb * c;
  }
  float scale = (j == 0 ? 1.0f : 2.0f) * (1.0f / 65536.0f);
  size_t o = (size_t)bo * 8192 + h * 32 + j;
  tre[o] = sr * scale;
  tim[o] = si * scale;
}

// ---- K6: inverse w-DFT via WMMA bf16, writes final output ------------------
// rows = (b*32+o)*256+h = 131072; K = 32 (j), N = 256 (w) in 16 tiles.
__global__ __launch_bounds__(256) void k_inv_wdft(
    const float* __restrict__ tre, const float* __restrict__ tim,
    float* __restrict__ out,
    const __bf16* __restrict__ bcos, const __bf16* __restrict__ bmsin) {
  const int lane = threadIdx.x & 31;
  const int wave = (blockIdx.x << 3) + (threadIdx.x >> 5);
  const int r0 = wave << 4;
  const int lr = lane & 15, hi = lane >> 4;
  const float* pr = tre + (size_t)(r0 + lr) * 32 + hi * 8;
  const float* pq = tim + (size_t)(r0 + lr) * 32 + hi * 8;
  v16bf aRe, aIm;
  {
    f32x4 q0 = *(const f32x4*)(pr),      q1 = *(const f32x4*)(pr + 4);
    f32x4 q2 = *(const f32x4*)(pr + 16), q3 = *(const f32x4*)(pr + 20);
    f32x4 u0 = *(const f32x4*)(pq),      u1 = *(const f32x4*)(pq + 4);
    f32x4 u2 = *(const f32x4*)(pq + 16), u3 = *(const f32x4*)(pq + 20);
#pragma unroll
    for (int e = 0; e < 4; ++e) {
      aRe[e] = (__bf16)q0[e]; aRe[e + 4] = (__bf16)q1[e];
      aRe[e + 8] = (__bf16)q2[e]; aRe[e + 12] = (__bf16)q3[e];
      aIm[e] = (__bf16)u0[e]; aIm[e + 4] = (__bf16)u1[e];
      aIm[e + 8] = (__bf16)u2[e]; aIm[e + 12] = (__bf16)u3[e];
    }
  }
  const int col = lane & 15;
#pragma unroll
  for (int nt = 0; nt < 16; ++nt) {
    v8f acc = {0,0,0,0,0,0,0,0};
    const v16bf bc = *(const v16bf*)(bcos  + ((nt * 32 + lane) << 4));
    const v16bf bm = *(const v16bf*)(bmsin + ((nt * 32 + lane) << 4));
    acc = wmma_bf16(aIm, bm, acc);
    acc = wmma_bf16(aRe, bc, acc);
#pragma unroll
    for (int r = 0; r < 8; ++r)
      out[(size_t)(r0 + r + hi * 8) * 256 + nt * 16 + col] = acc[r];
  }
}

// ---------------------------------------------------------------------------
extern "C" void kernel_launch(void* const* d_in, const int* in_sizes, int n_in,
                              void* d_out, int out_size, void* d_ws, size_t ws_size,
                              hipStream_t stream) {
  const float* x    = (const float*)d_in[0];  // (16,32,256,256)
  const float* cre  = (const float*)d_in[1];  // (32,32,32,32)
  const float* cim  = (const float*)d_in[2];
  const float* proj = (const float*)d_in[3];  // (32,32)
  float* out = (float*)d_out;                 // (16,32,256,256)

  char* ws = (char*)d_ws;
  float*  twc   = (float*)(ws + OFF_TWH_COS);
  float*  tws   = (float*)(ws + OFF_TWH_SIN);
  __bf16* fbc   = (__bf16*)(ws + OFF_FWDB_COS);
  __bf16* fbs   = (__bf16*)(ws + OFF_FWDB_SIN);
  __bf16* ibc   = (__bf16*)(ws + OFF_INVB_COS);
  __bf16* ibm   = (__bf16*)(ws + OFF_INVB_MSIN);
  float*  wre   = (float*)(ws + OFF_WT_RE);
  float*  wim   = (float*)(ws + OFF_WT_IM);
  float*  yre   = (float*)(ws + OFF_Y_RE);   // reused as Tre after K3 consumes Y
  float*  yim   = (float*)(ws + OFF_Y_IM);   // reused as Tim
  float*  xre   = (float*)(ws + OFF_X_RE);
  float*  xim   = (float*)(ws + OFF_X_IM);
  float*  gre   = (float*)(ws + OFF_G_RE);
  float*  gim   = (float*)(ws + OFF_G_IM);

  k_build_twh <<<32,    256, 0, stream>>>(twc, tws);
  k_build_fwdB<<<32,    256, 0, stream>>>(fbc, fbs);
  k_build_invB<<<32,    256, 0, stream>>>(ibc, ibm);
  k_wt        <<<4096,  256, 0, stream>>>(proj, cre, cim, wre, wim);
  k_fwd_wdft  <<<1024,  256, 0, stream>>>(x, yre, yim, fbc, fbs);
  k_fwd_hdft  <<<512,   256, 0, stream>>>(yre, yim, xre, xim, twc, tws);
  k_chanmix   <<<2048,  256, 0, stream>>>(xre, xim, wre, wim, gre, gim);
  k_inv_hdft  <<<16384, 256, 0, stream>>>(gre, gim, yre, yim, twc, tws);
  k_inv_wdft  <<<1024,  256, 0, stream>>>(yre, yim, out, ibc, ibm);
}
